// TangentManifoldVAE_5257039970412
// MI455X (gfx1250) — compile-verified, matrix operand order checked by reading, not run on hardware
//
#include <hip/hip_runtime.h>

// ---------------------------------------------------------------------------
// TangentManifoldVAE forward on gfx1250 (MI455X), bf16 WMMA + f32 accum.
// Attention is over a length-1 sequence => softmax == 1, so
//   z_att = (z @ a_wv + a_bv) @ a_wo + a_bo   (q/k projections are dead code)
// ---------------------------------------------------------------------------

typedef __attribute__((ext_vector_type(16))) __bf16 v16bf;
typedef __attribute__((ext_vector_type(8)))  float  v8f;

__device__ __forceinline__ unsigned short f32_to_bf16(float f) {
  unsigned int u = __float_as_uint(f);
  unsigned int r = (u + 0x7FFFu + ((u >> 16) & 1u)) >> 16;   // round-nearest-even
  return (unsigned short)r;
}
__device__ __forceinline__ unsigned pack_bf16x2(float lo, float hi) {
  return (unsigned)f32_to_bf16(lo) | ((unsigned)f32_to_bf16(hi) << 16);
}

union FragBF { uint4 u[2]; v16bf v; };

// ---- CDNA5 async global->LDS copy (ASYNCcnt-tracked), inline asm ----------
// global_load_async_to_lds_b128: VDST = LDS byte address (VGPR),
// VADDR = 64-bit global address (VGPR pair), saddr disabled ("off").
__device__ __forceinline__ void async_ld_b128(unsigned short* lds_dst,
                                              const unsigned short* gsrc) {
  asm volatile("global_load_async_to_lds_b128 %0, %1, off"
               :: "v"((unsigned)(uintptr_t)lds_dst), "v"(gsrc)
               : "memory");
}

__device__ __forceinline__ void wait_async0() {
#if __has_builtin(__builtin_amdgcn_s_wait_asynccnt)
  __builtin_amdgcn_s_wait_asynccnt(0);
#else
  asm volatile("s_wait_asynccnt 0x0" ::: "memory");
#endif
}

// ---------------------------------------------------------------------------
// GEMM:  C(f32[,bf16]) = epi( A_bf16[MxK] @ W_f32[KxN] + bias )
//   EPI: 0 = identity, 1 = tanh, 2 = clip(-10, 2);   EDGE: N % 64 != 0
// Block: 128x64 tile, BK=32, 8 waves, each wave 32x32 (2x2 WMMA frags).
// Double-buffered LDS; A tile via async-to-LDS DMA, W tile f32->bf16 packed
// into [n][k] (k-contiguous) so B-fragment loads are 2x contiguous b128.
// ---------------------------------------------------------------------------
template <int EPI, bool EDGE>
__global__ __launch_bounds__(256)
void gemm_wmma_bf16(const unsigned short* __restrict__ A,
                    const float* __restrict__ W,
                    const float* __restrict__ bias,
                    float* __restrict__ Cout,
                    unsigned short* __restrict__ Cbf,
                    int M, int N, int K) {
  (void)M;
  constexpr int BM = 128, BN = 64, BK = 32;
  __shared__ unsigned short As[2][BM * BK];   // [m][k], k contiguous
  __shared__ unsigned short Bs[2][BN * BK];   // [n][k], k contiguous

  const int tid  = threadIdx.x;
  const int lane = tid & 31;
  const int wave = tid >> 5;
  const int wm   = wave >> 1;                 // 0..3 (M)
  const int wn   = wave & 1;                  // 0..1 (N)
  const int bm0  = blockIdx.y * BM;
  const int bn0  = blockIdx.x * BN;
  const int kh   = lane >> 4;                 // K-half (fragment layout)
  const int l15  = lane & 15;

  // W-tile mapping: one (k-pair, 4n) unit per thread (exactly 256 units)
  const int kk = (tid >> 4) * 2;              // 0,2,...,30
  const int n4 = (tid & 15) * 4;              // 0..60
  const int gn = bn0 + n4;

  float va[4], vb[4];

  auto stageA = [&](int buf, int k0) {
#pragma unroll
    for (int i = 0; i < 2; ++i) {
      int c = tid + i * 256;                  // 0..511 x 16B chunks
      int m = c >> 2, seg = c & 3;
      async_ld_b128(&As[buf][m * BK + seg * 8],
                    A + (size_t)(bm0 + m) * K + k0 + seg * 8);
    }
  };
  auto loadB = [&](int k0) {
    const float* p0 = W + (size_t)(k0 + kk) * N + gn;
    const float* p1 = p0 + N;
    if (!EDGE) {
      float4 f0 = *reinterpret_cast<const float4*>(p0);
      float4 f1 = *reinterpret_cast<const float4*>(p1);
      va[0] = f0.x; va[1] = f0.y; va[2] = f0.z; va[3] = f0.w;
      vb[0] = f1.x; vb[1] = f1.y; vb[2] = f1.z; vb[3] = f1.w;
    } else {
#pragma unroll
      for (int j = 0; j < 4; ++j) {
        bool ok = (gn + j) < N;
        va[j] = ok ? p0[j] : 0.f;
        vb[j] = ok ? p1[j] : 0.f;
      }
    }
  };
  auto storeB = [&](int buf) {
#pragma unroll
    for (int j = 0; j < 4; ++j)
      *reinterpret_cast<unsigned*>(&Bs[buf][(n4 + j) * BK + kk]) =
          pack_bf16x2(va[j], vb[j]);
  };

  v8f acc[2][2] = {};

  // prologue: stage buffer 0
  stageA(0, 0);
  loadB(0);
  storeB(0);
  wait_async0();
  __syncthreads();

  for (int k0 = 0, p = 0; k0 < K; k0 += BK, p ^= 1) {
    const int  nk   = k0 + BK;
    const bool more = nk < K;
    if (more) { stageA(p ^ 1, nk); loadB(nk); }   // prefetch next tile

    FragBF af[2], bf[2];
#pragma unroll
    for (int mf = 0; mf < 2; ++mf) {
      const uint4* q = reinterpret_cast<const uint4*>(
          &As[p][(wm * 32 + mf * 16 + l15) * BK + kh * 16]);
      af[mf].u[0] = q[0]; af[mf].u[1] = q[1];
    }
#pragma unroll
    for (int nf = 0; nf < 2; ++nf) {
      const uint4* q = reinterpret_cast<const uint4*>(
          &Bs[p][(wn * 32 + nf * 16 + l15) * BK + kh * 16]);
      bf[nf].u[0] = q[0]; bf[nf].u[1] = q[1];
    }
#pragma unroll
    for (int mf = 0; mf < 2; ++mf)
#pragma unroll
      for (int nf = 0; nf < 2; ++nf)
        acc[mf][nf] = __builtin_amdgcn_wmma_f32_16x16x32_bf16(
            false, af[mf].v, false, bf[nf].v, (short)0, acc[mf][nf],
            false, false);

    if (more) { storeB(p ^ 1); wait_async0(); }
    __syncthreads();
  }

  // ---- epilogue ----
#pragma unroll
  for (int nf = 0; nf < 2; ++nf) {
    int col = bn0 + wn * 32 + nf * 16 + l15;
    if (EDGE && col >= N) continue;
    float bv = bias[col];
#pragma unroll
    for (int mf = 0; mf < 2; ++mf) {
#pragma unroll
      for (int r = 0; r < 8; ++r) {
        int row = bm0 + wm * 32 + mf * 16 + kh * 8 + r;
        float v = acc[mf][nf][r] + bv;
        if (EPI == 1) v = tanhf(v);
        if (EPI == 2) v = fminf(fmaxf(v, -10.f), 2.f);
        Cout[(size_t)row * N + col] = v;
        if (Cbf) Cbf[(size_t)row * N + col] = f32_to_bf16(v);
      }
    }
  }
}

// ---------------------------------------------------------------------------
// Fused LayerNorm + exact-erf GELU, one block per row, bf16 output.
// ---------------------------------------------------------------------------
__global__ __launch_bounds__(256)
void ln_gelu_kernel(const float* __restrict__ X,
                    const float* __restrict__ g,
                    const float* __restrict__ b,
                    unsigned short* __restrict__ Y, int N) {
  __shared__ float s1[256], s2[256];
  const int row = blockIdx.x;
  const int t = threadIdx.x;
  const float* x = X + (size_t)row * N;
  float sum = 0.f, sq = 0.f;
  for (int c = t; c < N; c += 256) { float v = x[c]; sum += v; sq += v * v; }
  s1[t] = sum; s2[t] = sq; __syncthreads();
  for (int o = 128; o; o >>= 1) {
    if (t < o) { s1[t] += s1[t + o]; s2[t] += s2[t + o]; }
    __syncthreads();
  }
  float mean = s1[0] / N;
  float var  = s2[0] / N - mean * mean;
  float inv  = rsqrtf(var + 1e-5f);
  for (int c = t; c < N; c += 256) {
    float y  = (x[c] - mean) * inv * g[c] + b[c];
    float ge = 0.5f * y * (1.0f + erff(y * 0.70710678118654752f));
    Y[(size_t)row * N + c] = f32_to_bf16(ge);
  }
}

__global__ void cvt_f32_bf16_kernel(const float* __restrict__ in,
                                    unsigned short* __restrict__ out, int n) {
  int i = blockIdx.x * blockDim.x + threadIdx.x;
  int stride = gridDim.x * blockDim.x;
  for (; i < n; i += stride) out[i] = f32_to_bf16(in[i]);
}

// ---- deterministic two-stage loss reductions ------------------------------
__global__ __launch_bounds__(256)
void sqdiff_partial_kernel(const float* __restrict__ a,
                           const float* __restrict__ b,
                           float* __restrict__ part, int n) {
  __shared__ float s[256];
  const int t = threadIdx.x;
  float acc = 0.f;
  for (int i = blockIdx.x * 256 + t; i < n; i += gridDim.x * 256) {
    float d = a[i] - b[i]; acc += d * d;
  }
  s[t] = acc; __syncthreads();
  for (int o = 128; o; o >>= 1) { if (t < o) s[t] += s[t + o]; __syncthreads(); }
  if (t == 0) part[blockIdx.x] = s[0];
}

__global__ __launch_bounds__(256)
void kl_partial_kernel(const float* __restrict__ mu,
                       const float* __restrict__ lv,
                       float* __restrict__ part, int n) {
  __shared__ float s[256];
  const int t = threadIdx.x;
  float acc = 0.f;
  for (int i = blockIdx.x * 256 + t; i < n; i += gridDim.x * 256) {
    float l = lv[i], m = mu[i];
    acc += 1.0f + l - m * m - expf(l);
  }
  s[t] = acc; __syncthreads();
  for (int o = 128; o; o >>= 1) { if (t < o) s[t] += s[t + o]; __syncthreads(); }
  if (t == 0) part[blockIdx.x] = s[0];
}

__global__ __launch_bounds__(256)
void finalize_kernel(const float* __restrict__ pr, const float* __restrict__ pk,
                     float* __restrict__ out3, float rscale, float kscale) {
  __shared__ float s1[256], s2[256];
  const int t = threadIdx.x;
  s1[t] = pr[t]; s2[t] = pk[t]; __syncthreads();
  for (int o = 128; o; o >>= 1) {
    if (t < o) { s1[t] += s1[t + o]; s2[t] += s2[t + o]; }
    __syncthreads();
  }
  if (t == 0) {
    out3[0] = s1[0] * rscale;                       // recon_loss
    out3[1] = fminf(-0.5f * s2[0] * kscale, 10.f);  // kl_loss (clamped)
    out3[2] = 0.f;                                  // reg_loss
  }
}

// ---------------------------------------------------------------------------
extern "C" void kernel_launch(void* const* d_in, const int* in_sizes, int n_in,
                              void* d_out, int out_size, void* d_ws, size_t ws_size,
                              hipStream_t stream) {
  (void)in_sizes; (void)n_in; (void)out_size; (void)ws_size;
  constexpr int B = 1024, D = 2048, H = 4096, L = 1024, C = 1000, H2 = 2048;

  const float* x     = (const float*)d_in[0];
  const float* e_w1  = (const float*)d_in[1];  const float* e_b1 = (const float*)d_in[2];
  const float* e_g1  = (const float*)d_in[3];  const float* e_be1= (const float*)d_in[4];
  const float* e_w2  = (const float*)d_in[5];  const float* e_b2 = (const float*)d_in[6];
  const float* e_g2  = (const float*)d_in[7];  const float* e_be2= (const float*)d_in[8];
  const float* mu_w  = (const float*)d_in[9];  const float* mu_b = (const float*)d_in[10];
  const float* var_w = (const float*)d_in[11]; const float* var_b= (const float*)d_in[12];
  // d_in[13..16] = a_wq, a_bq, a_wk, a_bk : dead (softmax over 1 key == 1)
  const float* a_wv  = (const float*)d_in[17]; const float* a_bv = (const float*)d_in[18];
  const float* a_wo  = (const float*)d_in[19]; const float* a_bo = (const float*)d_in[20];
  const float* d_w1  = (const float*)d_in[21]; const float* d_b1 = (const float*)d_in[22];
  const float* d_g1  = (const float*)d_in[23]; const float* d_be1= (const float*)d_in[24];
  const float* d_w2  = (const float*)d_in[25]; const float* d_b2 = (const float*)d_in[26];
  const float* d_g2  = (const float*)d_in[27]; const float* d_be2= (const float*)d_in[28];
  const float* d_w3  = (const float*)d_in[29]; const float* d_b3 = (const float*)d_in[30];
  const float* c_w1  = (const float*)d_in[31]; const float* c_b1 = (const float*)d_in[32];
  const float* c_g1  = (const float*)d_in[33]; const float* c_be1= (const float*)d_in[34];
  const float* c_w2  = (const float*)d_in[35]; const float* c_b2 = (const float*)d_in[36];
  const float* c_g2  = (const float*)d_in[37]; const float* c_be2= (const float*)d_in[38];
  const float* c_w3  = (const float*)d_in[39]; const float* c_b3 = (const float*)d_in[40];

  // ---- workspace carving --------------------------------------------------
  char* ws = (char*)d_ws;
  size_t off = 0;
  auto take = [&](size_t bytes) -> char* {
    char* p = ws + off;
    off = (off + bytes + 255) & ~(size_t)255;
    return p;
  };
  unsigned short* x_bf    = (unsigned short*)take((size_t)B * D * 2);
  float*          pre     = (float*)         take((size_t)B * H * 4);  // reused scratch
  float*          mu      = (float*)         take((size_t)B * L * 4);
  float*          logvar  = (float*)         take((size_t)B * L * 4);
  unsigned short* z_bf    = (unsigned short*)take((size_t)B * L * 2);
  unsigned short* v_bf    = (unsigned short*)take((size_t)B * L * 2);
  unsigned short* zatt_bf = (unsigned short*)take((size_t)B * L * 2);
  unsigned short* actA    = (unsigned short*)take((size_t)B * H * 2);
  unsigned short* actB    = (unsigned short*)take((size_t)B * H * 2);
  float*          partR   = (float*)         take(256 * 4);
  float*          partK   = (float*)         take(256 * 4);

  float* out_f   = (float*)d_out;
  float* logits  = out_f;                  // B x C
  float* x_recon = out_f + (size_t)B * C;  // B x D
  float* scalars = x_recon + (size_t)B * D;

  auto gemm = [&](const unsigned short* Abf, const float* Wt, const float* bias,
                  float* Cf, unsigned short* Cbf, int M, int N, int K,
                  int epi, bool edge) {
    dim3 grid((N + 63) / 64, M / 128), blk(256);
    if (edge)          gemm_wmma_bf16<0, true ><<<grid, blk, 0, stream>>>(Abf, Wt, bias, Cf, Cbf, M, N, K);
    else if (epi == 1) gemm_wmma_bf16<1, false><<<grid, blk, 0, stream>>>(Abf, Wt, bias, Cf, Cbf, M, N, K);
    else if (epi == 2) gemm_wmma_bf16<2, false><<<grid, blk, 0, stream>>>(Abf, Wt, bias, Cf, Cbf, M, N, K);
    else               gemm_wmma_bf16<0, false><<<grid, blk, 0, stream>>>(Abf, Wt, bias, Cf, Cbf, M, N, K);
  };
  auto lng = [&](const float* X, const float* g, const float* b,
                 unsigned short* Y, int N) {
    ln_gelu_kernel<<<dim3(B), dim3(256), 0, stream>>>(X, g, b, Y, N);
  };

  // ---- encoder ------------------------------------------------------------
  cvt_f32_bf16_kernel<<<dim3(2048), dim3(256), 0, stream>>>(x, x_bf, B * D);
  gemm(x_bf, e_w1, e_b1, pre, nullptr, B, H, D, 0, false);
  lng(pre, e_g1, e_be1, actA, H);
  gemm(actA, e_w2, e_b2, pre, nullptr, B, H, H, 0, false);
  lng(pre, e_g2, e_be2, actB, H);
  gemm(actB, mu_w,  mu_b,  mu,     z_bf,    B, L, H, 0, false); // z = mu (eval)
  gemm(actB, var_w, var_b, logvar, nullptr, B, L, H, 2, false); // clip(-10, 2)

  // ---- attention (softmax over singleton == 1) ----------------------------
  gemm(z_bf, a_wv, a_bv, pre, v_bf,    B, L, L, 0, false);
  gemm(v_bf, a_wo, a_bo, pre, zatt_bf, B, L, L, 0, false);

  // ---- decoder ------------------------------------------------------------
  gemm(z_bf, d_w1, d_b1, pre, nullptr, B, H, L, 0, false);
  lng(pre, d_g1, d_be1, actA, H);
  gemm(actA, d_w2, d_b2, pre, nullptr, B, H, H, 0, false);
  lng(pre, d_g2, d_be2, actB, H);
  gemm(actB, d_w3, d_b3, x_recon, nullptr, B, D, H, 1, false);  // tanh

  // ---- classifier ---------------------------------------------------------
  gemm(zatt_bf, c_w1, c_b1, pre, nullptr, B, H, L, 0, false);
  lng(pre, c_g1, c_be1, actA, H);
  gemm(actA, c_w2, c_b2, pre, nullptr, B, H2, H, 0, false);
  lng(pre, c_g2, c_be2, actB, H2);
  gemm(actB, c_w3, c_b3, logits, nullptr, B, C, H2, 0, true);   // N=1000 edge

  // ---- losses -------------------------------------------------------------
  sqdiff_partial_kernel<<<dim3(256), dim3(256), 0, stream>>>(x_recon, x, partR, B * D);
  kl_partial_kernel<<<dim3(256), dim3(256), 0, stream>>>(mu, logvar, partK, B * L);
  finalize_kernel<<<dim3(1), dim3(256), 0, stream>>>(
      partR, partK, scalars, 1.0f / (B * D), 1.0f / (B * L));
}
